// ViewpointLoss_38285338477081
// MI455X (gfx1250) — compile-verified, matrix operand order checked by reading, not run on hardware
//
#include <hip/hip_runtime.h>

typedef __attribute__((ext_vector_type(16))) _Float16 v16h;
typedef __attribute__((ext_vector_type(8)))  float    v8f;

#define NCLASS   12
#define PW       360            // CLASS_PERIOD
#define CP       (NCLASS * PW)  // 4320 floats per row
#define RPB      32             // rows per block (2 waves x 16 rows)
#define TPB      64             // 2 wave32 waves
#define LSTRIDE  361            // odd LDS row stride -> bank-conflict-free
#define NCHUNK   12             // ceil(360 / 32)

// ---- CDNA5 async global->LDS staging (ASYNCcnt path), guarded ----
#if defined(__has_builtin)
#if __has_builtin(__builtin_amdgcn_global_load_async_to_lds_b32) && \
    __has_builtin(__builtin_amdgcn_s_wait_asynccnt)
#define USE_ASYNC_LDS 1
#endif
#endif

#ifdef USE_ASYNC_LDS
typedef __attribute__((address_space(1))) int gint;   // global AS pointer elem
typedef __attribute__((address_space(3))) int lint;   // LDS AS pointer elem

__device__ __forceinline__ void async_copy_b32(const float* gsrc, float* ldst) {
    // AS(3) ptr value == 32-bit LDS offset == low 32 bits of the generic addr
    lint* l = (lint*)(unsigned int)(uintptr_t)ldst;
    gint* g = (gint*)(uintptr_t)gsrc;
    __builtin_amdgcn_global_load_async_to_lds_b32(g, l, 0, 0);
}
#endif

__global__ __launch_bounds__(TPB)
void vp_loss_main(const float* __restrict__ preds,
                  const float* __restrict__ labels,
                  const int*   __restrict__ objc,
                  float* __restrict__ part)
{
    __shared__ float ldsP[RPB * LSTRIDE];   // 46208 B preds tile
    __shared__ int   baseS[RPB];
    __shared__ float normS[RPB];
    __shared__ float zS[RPB];
    __shared__ float bp[2];

    const int t    = threadIdx.x;
    const int lane = t & 31;
    const int w    = t >> 5;
    const int r0   = blockIdx.x * RPB;

    if (t < RPB) baseS[t] = objc[r0 + t] * PW;
    __syncthreads();

    // ---- stage preds windows into LDS ----
#ifdef USE_ASYNC_LDS
    // direct global->LDS DMA, no VGPR round-trip; tracked by ASYNCcnt
    for (int row = 0; row < RPB; ++row) {
        const float* src = preds + (size_t)(r0 + row) * CP + baseS[row];
        float*       dst = &ldsP[row * LSTRIDE];
        for (int col = t; col < PW; col += TPB)
            async_copy_b32(src + col, dst + col);
    }
    __builtin_amdgcn_s_wait_asynccnt(0);
    __syncthreads();
#else
    for (int row = 0; row < RPB; ++row) {
        const float4* src4 = reinterpret_cast<const float4*>(
            preds + (size_t)(r0 + row) * CP + baseS[row]);
        float* dst = &ldsP[row * LSTRIDE];
        for (int c4 = t; c4 < PW / 4; c4 += TPB) {
            float4 v = src4[c4];
            dst[4 * c4 + 0] = v.x; dst[4 * c4 + 1] = v.y;
            dst[4 * c4 + 2] = v.z; dst[4 * c4 + 3] = v.w;
        }
    }
    __syncthreads();
#endif

    const int m      = lane & 15;        // A-matrix row fed by this lane
    const int hi     = lane >> 4;        // K-half select of A layout
    const int lr     = w * 16 + m;       // local row in tile
    const int rowOff = lr * LSTRIDE;

    v16h ones;
    #pragma unroll
    for (int i = 0; i < 16; ++i) ones[i] = (_Float16)1.0f;

    // ---- pass 1: L1 norms, D = |A| x ones (row-sum on matrix unit) ----
    v8f acc = {};
    #pragma unroll
    for (int c = 0; c < NCHUNK; ++c) {
        const int  kb = c * 32 + hi * 8;
        const bool va = (kb + 7)  < PW;
        const bool vb = (kb + 23) < PW;
        v16h a;
        #pragma unroll
        for (int i = 0; i < 8; ++i) {
            float v = va ? __builtin_fabsf(ldsP[rowOff + kb + i]) : 0.0f;
            a[i] = (_Float16)v;
        }
        #pragma unroll
        for (int i = 0; i < 8; ++i) {
            float v = vb ? __builtin_fabsf(ldsP[rowOff + kb + 16 + i]) : 0.0f;
            a[8 + i] = (_Float16)v;
        }
        acc = __builtin_amdgcn_wmma_f32_16x16x32_f16(false, a, false, ones,
                                                     (short)0, acc, false, false);
    }
    // D[m][n] == rowsum(m) for all n: lane0 holds rows 0-7, lane16 rows 8-15
    if (lane == 0) {
        #pragma unroll
        for (int r = 0; r < 8; ++r) normS[w * 16 + r] = acc[r];
    }
    if (lane == 16) {
        #pragma unroll
        for (int r = 0; r < 8; ++r) normS[w * 16 + 8 + r] = acc[r];
    }
    __syncthreads();
    const float inv = 1.0f / normS[lr];

    // ---- pass 2: row max of scaled values (lane pair splits row) ----
    float mx = -3.402823466e+38f;
    {
        const int c0 = hi * (PW / 2);
        for (int c = 0; c < PW / 2; ++c)
            mx = fmaxf(mx, ldsP[rowOff + c0 + c]);
    }
    mx = fmaxf(mx, __shfl_xor(mx, 16));
    mx *= inv;              // s > 0, so max(pw/s) = max(pw)/s

    // ---- pass 3: Z = sum exp(scaled - mx) via WMMA row-sum ----
    #pragma unroll
    for (int i = 0; i < 8; ++i) acc[i] = 0.0f;
    #pragma unroll
    for (int c = 0; c < NCHUNK; ++c) {
        const int  kb = c * 32 + hi * 8;
        const bool va = (kb + 7)  < PW;
        const bool vb = (kb + 23) < PW;
        v16h a;
        #pragma unroll
        for (int i = 0; i < 8; ++i) {
            float v = va ? __expf(ldsP[rowOff + kb + i] * inv - mx) : 0.0f;
            a[i] = (_Float16)v;
        }
        #pragma unroll
        for (int i = 0; i < 8; ++i) {
            float v = vb ? __expf(ldsP[rowOff + kb + 16 + i] * inv - mx) : 0.0f;
            a[8 + i] = (_Float16)v;
        }
        acc = __builtin_amdgcn_wmma_f32_16x16x32_f16(false, a, false, ones,
                                                     (short)0, acc, false, false);
    }
    if (lane == 0) {
        #pragma unroll
        for (int r = 0; r < 8; ++r) zS[w * 16 + r] = acc[r];
    }
    if (lane == 16) {
        #pragma unroll
        for (int r = 0; r < 8; ++r) zS[w * 16 + 8 + r] = acc[r];
    }
    __syncthreads();
    const float lz = __logf(zS[lr]);

    // ---- pass 4: sum lw * (scaled - mx - lz) via WMMA row-sum ----
    const float4* lab4 = reinterpret_cast<const float4*>(
        labels + (size_t)(r0 + lr) * CP + baseS[lr]);
    #pragma unroll
    for (int i = 0; i < 8; ++i) acc[i] = 0.0f;
    #pragma unroll
    for (int c = 0; c < NCHUNK; ++c) {
        const int  kb = c * 32 + hi * 8;
        const bool va = (kb + 7)  < PW;
        const bool vb = (kb + 23) < PW;
        const int  i0 = va ? (kb >> 2)        : 0;   // clamp: always in-bounds
        const int  i1 = vb ? ((kb + 16) >> 2) : 0;
        float4 l0 = lab4[i0];     float4 l1 = lab4[i0 + 1];
        float4 l2 = lab4[i1];     float4 l3 = lab4[i1 + 1];
        const float lw[16] = { l0.x, l0.y, l0.z, l0.w,  l1.x, l1.y, l1.z, l1.w,
                               l2.x, l2.y, l2.z, l2.w,  l3.x, l3.y, l3.z, l3.w };
        v16h a;
        #pragma unroll
        for (int i = 0; i < 8; ++i) {
            float ls = va ? (ldsP[rowOff + kb + i] * inv - mx - lz) : 0.0f;
            a[i] = (_Float16)(lw[i] * ls);
        }
        #pragma unroll
        for (int i = 0; i < 8; ++i) {
            float ls = vb ? (ldsP[rowOff + kb + 16 + i] * inv - mx - lz) : 0.0f;
            a[8 + i] = (_Float16)(lw[8 + i] * ls);
        }
        acc = __builtin_amdgcn_wmma_f32_16x16x32_f16(false, a, false, ones,
                                                     (short)0, acc, false, false);
    }
    // fold the wave-tile: lanes<16 hold rows 0-7 sums, lanes>=16 rows 8-15
    float tsum = 0.0f;
    #pragma unroll
    for (int r = 0; r < 8; ++r) tsum += acc[r];
    tsum += __shfl_xor(tsum, 16);
    if (lane == 0) bp[w] = tsum;
    __syncthreads();
    if (t == 0) part[blockIdx.x] = bp[0] + bp[1];
}

__global__ __launch_bounds__(256)
void vp_loss_reduce(const float* __restrict__ part, float* __restrict__ out, int n)
{
    __shared__ float s[256];
    const int t = threadIdx.x;
    float v = 0.0f;
    for (int i = t; i < n; i += 256) v += part[i];
    s[t] = v;
    __syncthreads();
    for (int off = 128; off > 0; off >>= 1) {
        if (t < off) s[t] += s[t + off];
        __syncthreads();
    }
    if (t == 0) out[0] = -s[0];
}

extern "C" void kernel_launch(void* const* d_in, const int* in_sizes, int n_in,
                              void* d_out, int out_size, void* d_ws, size_t ws_size,
                              hipStream_t stream) {
    const float* preds  = (const float*)d_in[0];
    const float* labels = (const float*)d_in[1];
    const int*   objc   = (const int*)d_in[2];
    float*       out    = (float*)d_out;
    float*       part   = (float*)d_ws;

    const int B = in_sizes[2];          // 16384 rows
    const int nblocks = B / RPB;        // 512

    vp_loss_main<<<nblocks, TPB, 0, stream>>>(preds, labels, objc, part);
    vp_loss_reduce<<<1, 256, 0, stream>>>(part, out, nblocks);
}
